// Model_41901700940060
// MI455X (gfx1250) — matrix operand order repaired and checked by reading, not compiled, not run on hardware
//
#include <hip/hip_runtime.h>
#include <hip/hip_bf16.h>

// ---------------------------------------------------------------------------
// GAT (3 blocks, 8 heads, F=1040) for MI455X / gfx1250.
// GEMMs run on v_wmma_f32_16x16x32_bf16 with double-buffered LDS tiles fed by
// GLOBAL_LOAD_ASYNC_TO_LDS_B128 (ASYNCcnt-tracked DMA, no VGPR round trip).
// Edge softmax/aggregation is the HBM-bound part and uses coalesced fp32
// passes + hardware float atomics.
// ---------------------------------------------------------------------------

#define FEAT   1040
#define KPAD   1056      // 33 * 32
#define NPAD   1088      // 17 * 64
#define NHEAD  8
#define DHEAD  130
#define NG     64
#define ASTR   40        // LDS row stride (bf16 elems) for A tile, bank-staggered
#define BSTR   40        // LDS row stride for Bt tile

typedef __bf16 v16bf __attribute__((ext_vector_type(16)));
typedef float  v8f   __attribute__((ext_vector_type(8)));

__device__ __forceinline__ unsigned short f2bf(float f) {
  unsigned u = __float_as_uint(f);
  unsigned r = ((u >> 16) & 1u) + 0x7FFFu;    // round-to-nearest-even
  return (unsigned short)((u + r) >> 16);
}
// monotone float<->uint mapping so unsigned atomicMax == float max
__device__ __forceinline__ unsigned f32_ord(float f) {
  unsigned b = __float_as_uint(f);
  return (b & 0x80000000u) ? ~b : (b | 0x80000000u);
}
__device__ __forceinline__ float ord_f32(unsigned u) {
  unsigned b = (u & 0x80000000u) ? (u ^ 0x80000000u) : ~u;
  return __uint_as_float(b);
}

// async DMA: 16 bytes global -> LDS, tracked by ASYNCcnt (gfx1250)
__device__ __forceinline__ void async_cp16(unsigned lds_off, const void* gaddr) {
  asm volatile("global_load_async_to_lds_b128 %0, %1, off"
               :: "v"(lds_off), "v"(gaddr)
               : "memory");
}
__device__ __forceinline__ void wait_async_le3() {
  asm volatile("s_wait_asynccnt 0x3" ::: "memory");
}
__device__ __forceinline__ void wait_async_0() {
  asm volatile("s_wait_asynccnt 0x0" ::: "memory");
}

// ---------------------------------------------------------------------------
__global__ void fill_u32_kernel(unsigned* __restrict__ p, unsigned v, long long n) {
  long long i = (long long)blockIdx.x * blockDim.x + threadIdx.x;
  if (i < n) p[i] = v;
}

// fp32 [N,FEAT] -> bf16 [Mpad,KPAD] (zero padded)
__global__ void cvt_act_kernel(const float* __restrict__ src,
                               unsigned short* __restrict__ dst,
                               int N, int Mpad) {
  long long i = (long long)blockIdx.x * blockDim.x + threadIdx.x;
  long long total = (long long)Mpad * KPAD;
  if (i >= total) return;
  int row = (int)(i / KPAD), k = (int)(i % KPAD);
  float v = (row < N && k < FEAT) ? src[(long long)row * FEAT + k] : 0.0f;
  dst[i] = f2bf(v);
}

// fp32 W [K=FEAT rows, FEAT cols] -> bf16 transposed [NPAD rows (out col), KPAD]
__global__ void cvt_w_kernel(const float* __restrict__ src,
                             unsigned short* __restrict__ dst) {
  long long i = (long long)blockIdx.x * blockDim.x + threadIdx.x;
  long long total = (long long)NPAD * KPAD;
  if (i >= total) return;
  int c = (int)(i / KPAD), k = (int)(i % KPAD);
  float v = (c < FEAT && k < FEAT) ? src[(long long)k * FEAT + c] : 0.0f;
  dst[i] = f2bf(v);
}

// ---------------------------------------------------------------------------
// C[M,FEAT] (f32) = A[Mpad,KPAD] (bf16) * B (bf16, stored transposed:
// Bt[NPAD,KPAD], row = output column, K contiguous).
// Workgroup tile 128x64, 8 waves, each wave: 16(M) x 64(N) via 4 WMMAs/K-step.
// LDS double-buffered; tiles land via async DMA while WMMAs run.
__global__ __launch_bounds__(256)
void gemm_bf16_wmma_kernel(const unsigned short* __restrict__ A,
                           const unsigned short* __restrict__ Bt,
                           float* __restrict__ C, int M) {
  __shared__ __align__(16) unsigned short As[2][128 * ASTR];
  __shared__ __align__(16) unsigned short Bs[2][64 * BSTR];

  const int tid  = threadIdx.x;
  const int lane = tid & 31;
  const int wave = tid >> 5;
  const int nBase = blockIdx.x * 64;
  const int mBase = blockIdx.y * 128;

  v8f acc[4] = {};

  // A-fragment addressing (ISA 7.12.2, 16-bit A 16x32): lane row = lane%16,
  // lanes>=16 read K offset +8; VGPR v holds K pair (v&3)*2 + (v>=4)*16.
  const int rA     = wave * 16 + (lane & 15);
  const int kHalfA = (lane >= 16) ? 8 : 0;
  // B-fragment: col = lane%16, lanes>=16 read K offset +8; VGPR v -> K pair (v&3)*2 + (v>=4)*16.
  const int cB     = lane & 15;
  const int kHalfB = (lane >= 16) ? 8 : 0;

  // issue one K-tile's DMA: per thread 2x16B of A (128x32) + 1x16B of B (64x32)
  auto issueTile = [&](int kk, int buf) {
#pragma unroll
    for (int it = 0; it < 2; ++it) {
      int idx = tid + it * 256;              // 0..511 chunks of 8 bf16
      int r = idx >> 2, q = idx & 3;
      unsigned loff = (unsigned)(size_t)(const void*)&As[buf][r * ASTR + q * 8];
      async_cp16(loff, A + (long long)(mBase + r) * KPAD + kk + q * 8);
    }
    {
      int r = tid >> 2, q = tid & 3;
      unsigned loff = (unsigned)(size_t)(const void*)&Bs[buf][r * BSTR + q * 8];
      async_cp16(loff, Bt + (long long)(nBase + r) * KPAD + kk + q * 8);
    }
  };

  const int T = KPAD / 32;
  issueTile(0, 0);

  for (int t = 0; t < T; ++t) {
    const int buf = t & 1;
    if (t + 1 < T) {
      issueTile((t + 1) * 32, buf ^ 1);  // next tile DMA in flight
      wait_async_le3();                  // retire current tile (3 ops/wave each)
    } else {
      wait_async_0();
    }
    __syncthreads();                     // everyone's tile t is in LDS

    // pack A fragment from LDS
    union { v16bf v; unsigned u[8]; } fa;
#pragma unroll
    for (int v = 0; v < 8; ++v) {
      int k = ((v & 3) * 2) + ((v >> 2) * 16) + kHalfA;
      fa.u[v] = *(const unsigned*)(&As[buf][rA * ASTR + k]);
    }
#pragma unroll
    for (int nt = 0; nt < 4; ++nt) {
      union { v16bf v; unsigned u[8]; } fb;
#pragma unroll
      for (int v = 0; v < 8; ++v) {
        int k = ((v & 3) * 2) + ((v >> 2) * 16) + kHalfB;
        fb.u[v] = *(const unsigned*)(&Bs[buf][(nt * 16 + cB) * BSTR + k]);
      }
      acc[nt] = __builtin_amdgcn_wmma_f32_16x16x32_bf16(
          false, fa.v, false, fb.v, (short)0, acc[nt], false, false);
    }
    __syncthreads();                     // done reading buf before its next DMA
  }

  // store C: VGPR i -> row (i + 8*(lane>=16)), col = nBase + nt*16 + lane%16
  const int rOff = (lane >= 16) ? 8 : 0;
#pragma unroll
  for (int nt = 0; nt < 4; ++nt) {
    int col = nBase + nt * 16 + cB;
    if (col >= FEAT) continue;
#pragma unroll
    for (int i = 0; i < 8; ++i) {
      int row = mBase + wave * 16 + i + rOff;
      if (row < M) C[(long long)row * FEAT + col] = acc[nt][i];
    }
  }
}

// ---------------------------------------------------------------------------
// per-node attention logits: asrc[n,h] = xh[n,h,:] . a_src[h,:], same for dst
__global__ void alpha_kernel(const float* __restrict__ xh,
                             const float* __restrict__ a_s,
                             const float* __restrict__ a_d,
                             float* __restrict__ asrc, float* __restrict__ adst,
                             int N) {
  int i = blockIdx.x * blockDim.x + threadIdx.x;
  if (i >= N * NHEAD) return;
  int n = i >> 3, h = i & 7;
  const float* xr = xh + (long long)n * FEAT + h * DHEAD;
  const float* as = a_s + h * DHEAD;
  const float* ad = a_d + h * DHEAD;
  float s = 0.f, d = 0.f;
#pragma unroll 2
  for (int j = 0; j < DHEAD; ++j) { float v = xr[j]; s += v * as[j]; d += v * ad[j]; }
  asrc[i] = s; adst[i] = d;
}

__device__ __forceinline__ void edge_sd(const int* ei, int E, int e, int& s, int& d) {
  if (e < E) { s = ei[e]; d = ei[E + e]; } else { s = d = e - E; }  // self loops
}

// pass1: e = leakyrelu(asrc[s]+adst[d]); segment max via ordered-uint atomicMax
__global__ void edge_pass1_kernel(const int* __restrict__ ei, int E, int E2,
                                  const float* __restrict__ asrc,
                                  const float* __restrict__ adst,
                                  float* __restrict__ ebuf,
                                  unsigned* __restrict__ emax) {
  int i = blockIdx.x * blockDim.x + threadIdx.x;
  if (i >= E2 * NHEAD) return;
  int e = i >> 3, h = i & 7, s, d;
  edge_sd(ei, E, e, s, d);
  float v = asrc[s * NHEAD + h] + adst[d * NHEAD + h];
  v = (v > 0.f) ? v : 0.2f * v;
  ebuf[i] = v;
  atomicMax(&emax[d * NHEAD + h], f32_ord(v));
}

// pass2: ex = exp(e - max); segment sum of ex
__global__ void edge_pass2_kernel(const int* __restrict__ ei, int E, int E2,
                                  float* __restrict__ ebuf,
                                  const unsigned* __restrict__ emax,
                                  float* __restrict__ denom) {
  int i = blockIdx.x * blockDim.x + threadIdx.x;
  if (i >= E2 * NHEAD) return;
  int e = i >> 3, h = i & 7, s, d;
  edge_sd(ei, E, e, s, d);
  float ex = __expf(ebuf[i] - ord_f32(emax[d * NHEAD + h]));
  ebuf[i] = ex;
  atomicAdd(&denom[d * NHEAD + h], ex);
}

// pass3: att[d,f] += (ex/denom[d,h]) * xh[s,f]   (grid-stride over E2*FEAT)
__global__ void edge_pass3_kernel(const int* __restrict__ ei, int E, int E2,
                                  const float* __restrict__ ebuf,
                                  const float* __restrict__ denom,
                                  const float* __restrict__ xh,
                                  float* __restrict__ att) {
  long long total = (long long)E2 * FEAT;
  long long stride = (long long)gridDim.x * blockDim.x;
  for (long long idx = (long long)blockIdx.x * blockDim.x + threadIdx.x;
       idx < total; idx += stride) {
    int e = (int)(idx / FEAT);
    int f = (int)(idx - (long long)e * FEAT);
    int h = f / DHEAD;
    int s, d;
    edge_sd(ei, E, e, s, d);
    float w = ebuf[e * NHEAD + h] / denom[d * NHEAD + h];
    atomicAdd(&att[(long long)d * FEAT + f], w * xh[(long long)s * FEAT + f]);
  }
}

// z = prev + att + b_att; BN(eval) ; ReLU
__global__ void bn_relu_kernel(const float* __restrict__ prev, int hasPrev,
                               const float* __restrict__ att,
                               const float* __restrict__ b_att,
                               const float* __restrict__ gamma,
                               const float* __restrict__ beta,
                               const float* __restrict__ mean,
                               const float* __restrict__ var,
                               float* __restrict__ out, int N) {
  long long idx = (long long)blockIdx.x * blockDim.x + threadIdx.x;
  if (idx >= (long long)N * FEAT) return;
  int f = (int)(idx % FEAT);
  float z = (hasPrev ? prev[idx] : 0.f) + att[idx] + b_att[f];
  z = (z - mean[f]) * rsqrtf(var[f] + 1e-5f) * gamma[f] + beta[f];
  out[idx] = (z > 0.f) ? z : 0.f;
}

__global__ void pool_count_kernel(const int* __restrict__ batch,
                                  float* __restrict__ counts, int N) {
  int i = blockIdx.x * blockDim.x + threadIdx.x;
  if (i < N) atomicAdd(&counts[batch[i]], 1.0f);
}

__global__ void pool_sum_kernel(const float* __restrict__ h,
                                const int* __restrict__ batch,
                                float* __restrict__ pooled, int N) {
  long long idx = (long long)blockIdx.x * blockDim.x + threadIdx.x;
  if (idx >= (long long)N * FEAT) return;
  int n = (int)(idx / FEAT), f = (int)(idx % FEAT);
  atomicAdd(&pooled[(long long)batch[n] * FEAT + f], h[idx]);
}

__global__ void final_out_kernel(const float* __restrict__ pooled,
                                 const float* __restrict__ counts,
                                 const float* __restrict__ Wout,
                                 const float* __restrict__ bout,
                                 float* __restrict__ out) {
  int i = threadIdx.x;
  if (i >= NG * 2) return;
  int g = i >> 1, c = i & 1;
  float inv = 1.0f / fmaxf(counts[g], 1.0f);
  float acc = 0.f;
  for (int f = 0; f < FEAT; ++f)
    acc += pooled[(long long)g * FEAT + f] * inv * Wout[f * 2 + c];
  out[i] = acc + bout[c];
}

// ---------------------------------------------------------------------------
static inline int blks(long long n) { return (int)((n + 255) / 256); }

extern "C" void kernel_launch(void* const* d_in, const int* in_sizes, int n_in,
                              void* d_out, int out_size, void* d_ws, size_t ws_size,
                              hipStream_t stream) {
  const float* x     = (const float*)d_in[0];
  const int*   ei    = (const int*)d_in[1];
  const int*   batch = (const int*)d_in[2];
  const float* W     = (const float*)d_in[3];
  const float* a_src = (const float*)d_in[4];
  const float* a_dst = (const float*)d_in[5];
  const float* b_att = (const float*)d_in[6];
  const float* gamma = (const float*)d_in[7];
  const float* beta  = (const float*)d_in[8];
  const float* rmean = (const float*)d_in[9];
  const float* rvar  = (const float*)d_in[10];
  const float* Wout  = (const float*)d_in[11];
  const float* bout  = (const float*)d_in[12];

  const int N  = in_sizes[0] / FEAT;      // 30000
  const int E  = in_sizes[1] / 2;         // 240000
  const int E2 = E + N;                   // with self loops
  const int Mpad = ((N + 127) / 128) * 128;

  // workspace carve-up (256B aligned)
  char* ws = (char*)d_ws;
  size_t off = 0;
  auto take = [&](size_t bytes) -> char* {
    char* p = ws + off; off += (bytes + 255) & ~(size_t)255; return p;
  };
  unsigned short* hbf  = (unsigned short*)take((size_t)Mpad * KPAD * 2);
  unsigned short* wbt  = (unsigned short*)take((size_t)NPAD * KPAD * 2);
  float* xh    = (float*)take((size_t)N * FEAT * 4);
  float* att   = (float*)take((size_t)N * FEAT * 4);
  float* bufA  = (float*)take((size_t)N * FEAT * 4);
  float* bufB  = (float*)take((size_t)N * FEAT * 4);
  float* bufC  = (float*)take((size_t)N * FEAT * 4);
  float* ebuf  = (float*)take((size_t)E2 * NHEAD * 4);
  float* asrc  = (float*)take((size_t)N * NHEAD * 4);
  float* adst  = (float*)take((size_t)N * NHEAD * 4);
  unsigned* emax = (unsigned*)take((size_t)N * NHEAD * 4);
  float* denom = (float*)take((size_t)N * NHEAD * 4);
  float* pooled = (float*)take((size_t)NG * FEAT * 4);
  float* counts = (float*)take((size_t)NG * 4);

  const float* h_in = x;
  const float* prev = nullptr;
  float* outs[3] = {bufA, bufB, bufC};

  for (int l = 0; l < 3; ++l) {
    // bf16 conversion (activations padded to Mpad x KPAD; weights transposed)
    cvt_act_kernel<<<blks((long long)Mpad * KPAD), 256, 0, stream>>>(h_in, hbf, N, Mpad);
    cvt_w_kernel<<<blks((long long)NPAD * KPAD), 256, 0, stream>>>(
        W + (size_t)l * FEAT * FEAT, wbt);

    // WMMA GEMM: xh = h * W[l]
    gemm_bf16_wmma_kernel<<<dim3(NPAD / 64, Mpad / 128), 256, 0, stream>>>(hbf, wbt, xh, N);

    // attention logits
    alpha_kernel<<<blks((long long)N * NHEAD), 256, 0, stream>>>(
        xh, a_src + (size_t)l * NHEAD * DHEAD, a_dst + (size_t)l * NHEAD * DHEAD,
        asrc, adst, N);

    // zero accumulators
    fill_u32_kernel<<<blks((long long)N * FEAT), 256, 0, stream>>>((unsigned*)att, 0u, (long long)N * FEAT);
    fill_u32_kernel<<<blks((long long)N * NHEAD), 256, 0, stream>>>(emax, 0u, (long long)N * NHEAD);
    fill_u32_kernel<<<blks((long long)N * NHEAD), 256, 0, stream>>>((unsigned*)denom, 0u, (long long)N * NHEAD);

    // segment softmax over incoming edges + weighted aggregation
    edge_pass1_kernel<<<blks((long long)E2 * NHEAD), 256, 0, stream>>>(ei, E, E2, asrc, adst, ebuf, emax);
    edge_pass2_kernel<<<blks((long long)E2 * NHEAD), 256, 0, stream>>>(ei, E, E2, ebuf, emax, denom);
    edge_pass3_kernel<<<131072, 256, 0, stream>>>(ei, E, E2, ebuf, denom, xh, att);

    // residual + BN(eval) + ReLU
    bn_relu_kernel<<<blks((long long)N * FEAT), 256, 0, stream>>>(
        prev, prev != nullptr, att, b_att + (size_t)l * FEAT,
        gamma + (size_t)l * FEAT, beta + (size_t)l * FEAT,
        rmean + (size_t)l * FEAT, rvar + (size_t)l * FEAT, outs[l], N);

    prev = h_in;
    h_in = outs[l];
  }

  // global mean pool + linear head
  fill_u32_kernel<<<blks((long long)NG * FEAT), 256, 0, stream>>>((unsigned*)pooled, 0u, (long long)NG * FEAT);
  fill_u32_kernel<<<1, 256, 0, stream>>>((unsigned*)counts, 0u, NG);
  pool_count_kernel<<<blks(N), 256, 0, stream>>>(batch, counts, N);
  pool_sum_kernel<<<blks((long long)N * FEAT), 256, 0, stream>>>(h_in, batch, pooled, N);
  final_out_kernel<<<1, 128, 0, stream>>>(pooled, counts, Wout, bout, (float*)d_out);
}
